// FourierBaseDynamic_5686536700280
// MI455X (gfx1250) — compile-verified, matrix operand order checked by reading, not run on hardware
//
#include <hip/hip_runtime.h>

typedef __attribute__((ext_vector_type(16))) _Float16 v16h;
typedef __attribute__((ext_vector_type(8)))  _Float16 v8h;
typedef __attribute__((ext_vector_type(8)))  float    v8f;
typedef __attribute__((ext_vector_type(4)))  float    v4f;

#define E_TOT   200000
#define D_DIM   64
#define F_DIM   32
#define K1_BLK  130          // 128 bilinear K-blocks + 2 bias K-blocks
#define K2_BLK  8            // 256 / 32 (feat dim padded 6 -> 8)
#define FRAG_ELEMS 512       // 32 lanes * 16 f16 per B fragment
#define B1_FRAGS (K1_BLK*2)  // 260
#define B2_FRAGS (K2_BLK*2)  // 16
#define B2_ELEM_OFF (B1_FRAGS*FRAG_ELEMS)   // 133120 f16 elements
#define WAVES_PER_WG 3
#define EDGES_PER_WAVE 32
#define HS_STRIDE  72        // 144B row stride -> 36-bank step, conflict-free
#define EMB_STRIDE 264       // 528B row stride -> 4-bank step, conflict-free
#define WAVE_SMEM  (EDGES_PER_WAVE*EMB_STRIDE)   // 8448 f16 (hs region overlaid)

static __device__ __forceinline__ v8f wmma_f16(v16h a, v16h b, v8f c) {
  return __builtin_amdgcn_wmma_f32_16x16x32_f16(false, a, false, b,
                                                (short)0, c, false, false);
}

// ---------------------------------------------------------------------------
// Kernel 1: pack W_freq/b_freq (K=4160) and W_out (K2=256, emb order
// [f_emb*8 + feat], feat 6..7 zero) into f16 WMMA B-fragments, per-lane
// contiguous: lane<16 holds column f, K={0..7,16..23}; lane>=16 K={8..15,24..31}.
// ---------------------------------------------------------------------------
__global__ void pack_weights(const float* __restrict__ W_freq,
                             const float* __restrict__ b_freq,
                             const float* __restrict__ W_out,
                             _Float16* __restrict__ ws) {
  int tid = blockIdx.x * blockDim.x + threadIdx.x;
  const int total1 = B1_FRAGS * FRAG_ELEMS;
  const int total2 = B2_FRAGS * FRAG_ELEMS;
  if (tid < total1) {
    int frag = tid >> 9;
    int lane = (tid >> 4) & 31;
    int j    = tid & 15;
    int kb = frag >> 1, ft = frag & 1;
    int f  = ft * 16 + (lane & 15);
    int k  = (j < 8 ? j : j + 8) + ((lane >= 16) ? 8 : 0);   // 0..31 in-block K
    float v;
    if (kb < 128) {
      int d0   = kb >> 1;
      int kcol = ((kb & 1) << 5) + k;
      v = W_freq[(f * 64 + d0) * 64 + kcol];     // W2[f, kb*32+k]
    } else {
      int d = ((kb - 128) << 5) + k;
      v = b_freq[f * 64 + d];                    // bias K-extension
    }
    ws[tid] = (_Float16)v;
  } else if (tid < total1 + total2) {
    int t2   = tid - total1;
    int frag = t2 >> 9;
    int lane = (t2 >> 4) & 31;
    int j    = t2 & 15;
    int kb2 = frag >> 1, ft = frag & 1;
    int f  = ft * 16 + (lane & 15);
    int k  = (j < 8 ? j : j + 8) + ((lane >= 16) ? 8 : 0);
    int k2 = kb2 * 32 + k;            // 0..255 permuted emb index
    int f_emb = k2 >> 3;              // which Fourier frequency column
    int feat  = k2 & 7;               // sin1,cos1,sin2,cos2,sin4,cos4,pad,pad
    float v = (feat < 6) ? W_out[f * (6 * F_DIM) + feat * F_DIM + f_emb] : 0.0f;
    ws[tid] = (_Float16)v;
  }
}

// ---------------------------------------------------------------------------
// Kernel 2: fused bilinear-proj -> Fourier features -> out GEMM.
// One wave handles 32 edges (2 M-tiles), fully independent (no barriers).
// ---------------------------------------------------------------------------
__global__ __launch_bounds__(WAVES_PER_WG * 32)
void fourier_edge_kernel(const float* __restrict__ h_src,
                         const float* __restrict__ h_i,
                         const float* __restrict__ b_out,
                         const _Float16* __restrict__ wsB,
                         float* __restrict__ out) {
  __shared__ _Float16 smem[WAVES_PER_WG][WAVE_SMEM];   // 50688 B

  const int wave = threadIdx.x >> 5;
  const int lane = threadIdx.x & 31;
  const int el   = lane & 15;
  const int hi   = lane >> 4;           // 0/1: K-half of the wave
  const int base = hi << 3;             // 0 or 8
  const int e_base = blockIdx.x * (WAVES_PER_WG * EDGES_PER_WAVE)
                   + wave * EDGES_PER_WAVE;

  _Float16* hs = &smem[wave][0];        // [32][HS_STRIDE] view (overlaid)

  // --- stage h_src rows as f16 into LDS (one edge per lane) ---
  {
    int e = e_base + lane; if (e >= E_TOT) e = E_TOT - 1;
    const float* row = h_src + e * D_DIM;
    _Float16* dst = hs + lane * HS_STRIDE;
    #pragma unroll
    for (int d = 0; d < D_DIM; d += 4) {
      v4f v = *(const v4f*)(row + d);
      dst[d+0] = (_Float16)v[0]; dst[d+1] = (_Float16)v[1];
      dst[d+2] = (_Float16)v[2]; dst[d+3] = (_Float16)v[3];
    }
  }

  // --- per-lane h_i templates: T[mt][p] covers K-slices for k0 = p*32 ---
  v16h T[2][2];
  #pragma unroll
  for (int mt = 0; mt < 2; ++mt) {
    int e = e_base + mt * 16 + el; if (e >= E_TOT) e = E_TOT - 1;
    const float* row = h_i + e * D_DIM;
    #pragma unroll
    for (int p = 0; p < 2; ++p) {
      int c0 = p * 32 + base;
      v4f a0 = *(const v4f*)(row + c0);
      v4f a1 = *(const v4f*)(row + c0 + 4);
      v4f b0 = *(const v4f*)(row + c0 + 16);
      v4f b1 = *(const v4f*)(row + c0 + 20);
      v16h t;
      #pragma unroll
      for (int j = 0; j < 4; ++j) {
        t[j]    = (_Float16)a0[j]; t[4+j]  = (_Float16)a1[j];
        t[8+j]  = (_Float16)b0[j]; t[12+j] = (_Float16)b1[j];
      }
      T[mt][p] = t;
    }
  }

  // --- main K loop, software-pipelined A scaling (no WMMA WAR hazards) ---
  v8f c00{}, c01{}, c10{}, c11{};
  const _Float16* hs0 = hs + el * HS_STRIDE;
  const _Float16* hs1 = hs + (16 + el) * HS_STRIDE;
  v16h a0 = T[0][0] * hs0[0];
  v16h a1 = T[1][0] * hs1[0];
  #pragma unroll 2
  for (int kb = 0; kb < 128; ++kb) {
    v16h b0 = *(const v16h*)(wsB + (kb*2 + 0) * FRAG_ELEMS + lane * 16);
    v16h b1 = *(const v16h*)(wsB + (kb*2 + 1) * FRAG_ELEMS + lane * 16);
    int kbn = (kb + 1) & 127;           // wrap: last prefetch unused
    _Float16 s0 = hs0[kbn >> 1];
    _Float16 s1 = hs1[kbn >> 1];
    v16h a0n = T[0][kbn & 1] * s0;      // writes the *other* buffer ->
    v16h a1n = T[1][kbn & 1] * s1;      // co-executes with WMMAs below
    c00 = wmma_f16(a0, b0, c00);
    c01 = wmma_f16(a0, b1, c01);
    c10 = wmma_f16(a1, b0, c10);
    c11 = wmma_f16(a1, b1, c11);
    a0 = a0n; a1 = a1n;
  }
  // --- 2 bias K-blocks: A = raw h_src slices ---
  #pragma unroll
  for (int p = 0; p < 2; ++p) {
    int kb = 128 + p;
    v16h b0 = *(const v16h*)(wsB + (kb*2 + 0) * FRAG_ELEMS + lane * 16);
    v16h b1 = *(const v16h*)(wsB + (kb*2 + 1) * FRAG_ELEMS + lane * 16);
    v16h a[2];
    #pragma unroll
    for (int mt = 0; mt < 2; ++mt) {
      int e = e_base + mt * 16 + el; if (e >= E_TOT) e = E_TOT - 1;
      const float* row = h_src + e * D_DIM;
      int c0 = p * 32 + base;
      v4f x0 = *(const v4f*)(row + c0);
      v4f x1 = *(const v4f*)(row + c0 + 4);
      v4f y0 = *(const v4f*)(row + c0 + 16);
      v4f y1 = *(const v4f*)(row + c0 + 20);
      v16h t;
      #pragma unroll
      for (int j = 0; j < 4; ++j) {
        t[j]    = (_Float16)x0[j]; t[4+j]  = (_Float16)x1[j];
        t[8+j]  = (_Float16)y0[j]; t[12+j] = (_Float16)y1[j];
      }
      a[mt] = t;
    }
    c00 = wmma_f16(a[0], b0, c00);
    c01 = wmma_f16(a[0], b1, c01);
    c10 = wmma_f16(a[1], b0, c10);
    c11 = wmma_f16(a[1], b1, c11);
  }

  // --- Fourier features: 2 transcendentals + double-angle cascade,
  //     packed 16B LDS store per proj value ---
  _Float16* emb = &smem[wave][0];       // [32][EMB_STRIDE] view (overlays hs)
  {
    v8f cc[2][2] = {{c00, c01}, {c10, c11}};
    #pragma unroll
    for (int mt = 0; mt < 2; ++mt)
      #pragma unroll
      for (int ft = 0; ft < 2; ++ft) {
        int f = ft * 16 + el;
        #pragma unroll
        for (int r = 0; r < 8; ++r) {
          float p = cc[mt][ft][r];
          float s1 = __sinf(p), c1 = __cosf(p);
          float s2 = 2.0f * s1 * c1, c2 = 1.0f - 2.0f * s1 * s1;
          float s4 = 2.0f * s2 * c2, c4 = 1.0f - 2.0f * s2 * s2;
          v8h pk;
          pk[0] = (_Float16)s1; pk[1] = (_Float16)c1;
          pk[2] = (_Float16)s2; pk[3] = (_Float16)c2;
          pk[4] = (_Float16)s4; pk[5] = (_Float16)c4;
          pk[6] = (_Float16)0.0f; pk[7] = (_Float16)0.0f;
          int erow = mt * 16 + r + hi * 8;
          *(v8h*)(emb + erow * EMB_STRIDE + f * 8) = pk;
        }
      }
  }

  // --- output GEMM: emb[32,256] @ packed W_out^T, K2 = 8 blocks ---
  v8f o[2][2] = {};
  #pragma unroll
  for (int kb2 = 0; kb2 < K2_BLK; ++kb2) {
    v16h b0 = *(const v16h*)(wsB + B2_ELEM_OFF + (kb2*2 + 0) * FRAG_ELEMS + lane * 16);
    v16h b1 = *(const v16h*)(wsB + B2_ELEM_OFF + (kb2*2 + 1) * FRAG_ELEMS + lane * 16);
    #pragma unroll
    for (int mt = 0; mt < 2; ++mt) {
      const _Float16* rw = emb + (mt * 16 + el) * EMB_STRIDE;
      int c0 = kb2 * 32 + base;
      v8h lo = *(const v8h*)(rw + c0);
      v8h hv = *(const v8h*)(rw + c0 + 16);
      v16h a = __builtin_shufflevector(lo, hv, 0,1,2,3,4,5,6,7,
                                               8,9,10,11,12,13,14,15);
      o[mt][0] = wmma_f16(a, b0, o[mt][0]);
      o[mt][1] = wmma_f16(a, b1, o[mt][1]);
    }
  }

  // --- add b_out, store (guarded for the edge tail) ---
  float bo0 = b_out[el];
  float bo1 = b_out[16 + el];
  #pragma unroll
  for (int mt = 0; mt < 2; ++mt)
    #pragma unroll
    for (int ft = 0; ft < 2; ++ft) {
      float bo = ft ? bo1 : bo0;
      #pragma unroll
      for (int r = 0; r < 8; ++r) {
        int eg = e_base + mt * 16 + r + hi * 8;
        if (eg < E_TOT)
          out[eg * F_DIM + ft * 16 + el] = o[mt][ft][r] + bo;
      }
    }
}

// ---------------------------------------------------------------------------
extern "C" void kernel_launch(void* const* d_in, const int* in_sizes, int n_in,
                              void* d_out, int out_size, void* d_ws, size_t ws_size,
                              hipStream_t stream) {
  const float* h_src  = (const float*)d_in[0];
  const float* h_i    = (const float*)d_in[1];
  const float* W_freq = (const float*)d_in[2];
  const float* b_freq = (const float*)d_in[3];
  const float* W_out  = (const float*)d_in[4];
  const float* b_out  = (const float*)d_in[5];
  _Float16* ws  = (_Float16*)d_ws;
  float*    out = (float*)d_out;

  // 276 fragments * 512 f16 each = 141312 threads
  const int pack_threads = (B1_FRAGS + B2_FRAGS) * FRAG_ELEMS;
  pack_weights<<<(pack_threads + 255) / 256, 256, 0, stream>>>(W_freq, b_freq, W_out, ws);

  const int edges_per_wg = WAVES_PER_WG * EDGES_PER_WAVE;     // 96
  const int grid = (E_TOT + edges_per_wg - 1) / edges_per_wg; // 2084
  fourier_edge_kernel<<<grid, WAVES_PER_WG * 32, 0, stream>>>(h_src, h_i, b_out, ws, out);
}
